// GineXmlEncoder_73873437491714
// MI455X (gfx1250) — matrix operand order repaired and checked by reading, not compiled, not run on hardware
//
#include <hip/hip_runtime.h>
#include <hip/hip_bf16.h>
#include <stdint.h>

// ---------------------------------------------------------------------------
// CDNA5 (gfx1250) WMMA bf16 path. wave32. D = A(16x32) x B(32x16) + C(16x16 f32)
// ---------------------------------------------------------------------------
typedef __attribute__((ext_vector_type(16))) __bf16 v16bf;
typedef __attribute__((ext_vector_type(8)))  float  v8f;
typedef __attribute__((ext_vector_type(4)))  int    v4i;
typedef __attribute__((address_space(3)))    v4i    v4i_lds;

union FragU { v16bf v; uint4 q[2]; };

#if defined(__has_builtin)
# if __has_builtin(__builtin_amdgcn_global_load_async_to_lds_b128)
#  define HAVE_ASYNC_LDS 1
# endif
#endif
#ifndef HAVE_ASYNC_LDS
# define HAVE_ASYNC_LDS 0
#endif

// Copy 32 contiguous bytes global->LDS. On gfx1250 use the async-to-LDS path
// (ASYNCcnt tracked, bypasses VGPRs); otherwise a plain synchronous copy.
__device__ __forceinline__ void async_copy32(const __bf16* g, __bf16* l) {
#if HAVE_ASYNC_LDS
  v4i* gp = (v4i*)g;          // builtin arg0: v4i* (non-const)
  v4i_lds* lp = (v4i_lds*)l;  // builtin arg1: v4i __shared__*
  __builtin_amdgcn_global_load_async_to_lds_b128(gp, lp, 0, 0);
  __builtin_amdgcn_global_load_async_to_lds_b128(gp, lp, 16, 0);
#else
  ((uint4*)l)[0] = ((const uint4*)g)[0];
  ((uint4*)l)[1] = ((const uint4*)g)[1];
#endif
}

__device__ __forceinline__ void async_wait() {
#if HAVE_ASYNC_LDS
# if defined(__has_builtin) && __has_builtin(__builtin_amdgcn_s_wait_asynccnt)
  __builtin_amdgcn_s_wait_asynccnt(0);
# else
  asm volatile("s_wait_asynccnt 0x0" ::: "memory");
# endif
#endif
}

// A fragment (16x32 bf16): lane l -> row (l&15);
// holds K = kb..kb+15, kb = (l>>4)*16.
__device__ __forceinline__ v16bf load_a_frag(const __bf16* base, int rowBase,
                                             int stride, int kOff, int lane) {
  const int r  = rowBase + (lane & 15);
  const int kb = kOff + ((lane >> 4) << 4);
  FragU f;
  f.q[0] = *(const uint4*)(base + (size_t)r * stride + kb);
  f.q[1] = *(const uint4*)(base + (size_t)r * stride + kb + 8);
  return f.v;
}

// B fragment (32x16 bf16) from transposed weights Wt[Nout][K]:
// lane l -> column (l&15); holds K = kh..kh+15, kh = (l>>4)*16.
__device__ __forceinline__ v16bf load_b_frag(const __bf16* Wt, int K, int colBase,
                                             int k0, int lane, int Nout) {
  int col = colBase + (lane & 15);
  col = col < Nout ? col : Nout - 1;   // clamp: keep EXEC all-ones, guard stores
  const int kh = k0 + ((lane >> 4) << 4);
  const uint4* p = (const uint4*)(Wt + (size_t)col * K + kh);
  FragU f;
  f.q[0] = p[0];
  f.q[1] = p[1];
  return f.v;
}

__device__ __forceinline__ v8f wmma_bf16(v16bf a, v16bf b, v8f c) {
  return __builtin_amdgcn_wmma_f32_16x16x32_bf16(false, a, false, b,
                                                 (short)0, c, false, false);
}

// Convert 16 consecutive f32 -> 16 bf16, store 32B to LDS (two b128 stores).
__device__ __forceinline__ void cvt_store16(const float* s, __bf16* d) {
  float4 f0 = ((const float4*)s)[0];
  float4 f1 = ((const float4*)s)[1];
  float4 f2 = ((const float4*)s)[2];
  float4 f3 = ((const float4*)s)[3];
  union { __bf16 b[16]; uint4 q[2]; } pk;
  pk.b[0]=(__bf16)f0.x;  pk.b[1]=(__bf16)f0.y;  pk.b[2]=(__bf16)f0.z;  pk.b[3]=(__bf16)f0.w;
  pk.b[4]=(__bf16)f1.x;  pk.b[5]=(__bf16)f1.y;  pk.b[6]=(__bf16)f1.z;  pk.b[7]=(__bf16)f1.w;
  pk.b[8]=(__bf16)f2.x;  pk.b[9]=(__bf16)f2.y;  pk.b[10]=(__bf16)f2.z; pk.b[11]=(__bf16)f2.w;
  pk.b[12]=(__bf16)f3.x; pk.b[13]=(__bf16)f3.y; pk.b[14]=(__bf16)f3.z; pk.b[15]=(__bf16)f3.w;
  *(uint4*)d       = pk.q[0];
  *(uint4*)(d + 8) = pk.q[1];
}

// ---------------------------------------------------------------------------
// GEMM (fp32 A): C[M,Nout] = act(A@W (+ A2@W2) + bias)
// Block: 256 thr (8 waves), tile 128x64, BK=32. Weights bf16 transposed
// Wt[Nout][K]; fp32 A converted to bf16 while staging into LDS.
// ---------------------------------------------------------------------------
template<bool RELU, bool DUAL, bool OBF>
__global__ void __launch_bounds__(256, 2) gemm_wmma(
    const float* __restrict__ A, const float* __restrict__ A2, int lda,
    const __bf16* __restrict__ Wt, const __bf16* __restrict__ Wt2,
    const float* __restrict__ bias,
    void* __restrict__ C, int ldc, int M, int K, int Nout)
{
  __shared__ __bf16 As [128 * 32];
  __shared__ __bf16 As2[128 * 32];

  const int tid  = threadIdx.x;
  const int lane = tid & 31;
  const int wid  = tid >> 5;
  const int msub = (wid & 3) << 5;   // 0,32,64,96
  const int nsub = (wid >> 2) << 5;  // 0,32
  const int m0   = blockIdx.x * 128;
  const int n0   = blockIdx.y * 64;

  __builtin_prefetch(Wt + (size_t)(n0 + nsub + (lane & 15)) * K, 0, 1);

  v8f acc00 = {}; v8f acc01 = {}; v8f acc10 = {}; v8f acc11 = {};

  for (int k0 = 0; k0 < K; k0 += 32) {
    __syncthreads();
    {
      const int r  = tid >> 1;
      const int hh = (tid & 1) << 4;
      int rg = m0 + r; rg = rg < M ? rg : M - 1;
      cvt_store16(A + (size_t)rg * lda + k0 + hh, As + r * 32 + hh);
      if constexpr (DUAL) {
        cvt_store16(A2 + (size_t)rg * lda + k0 + hh, As2 + r * 32 + hh);
      }
    }
    __syncthreads();

    v16bf a0 = load_a_frag(As, msub,      32, 0, lane);
    v16bf a1 = load_a_frag(As, msub + 16, 32, 0, lane);
    v16bf b0 = load_b_frag(Wt, K, n0 + nsub,      k0, lane, Nout);
    v16bf b1 = load_b_frag(Wt, K, n0 + nsub + 16, k0, lane, Nout);
    acc00 = wmma_bf16(a0, b0, acc00);
    acc01 = wmma_bf16(a0, b1, acc01);
    acc10 = wmma_bf16(a1, b0, acc10);
    acc11 = wmma_bf16(a1, b1, acc11);
    if constexpr (DUAL) {
      v16bf c0 = load_a_frag(As2, msub,      32, 0, lane);
      v16bf c1 = load_a_frag(As2, msub + 16, 32, 0, lane);
      v16bf d0 = load_b_frag(Wt2, K, n0 + nsub,      k0, lane, Nout);
      v16bf d1 = load_b_frag(Wt2, K, n0 + nsub + 16, k0, lane, Nout);
      acc00 = wmma_bf16(c0, d0, acc00);
      acc01 = wmma_bf16(c0, d1, acc01);
      acc10 = wmma_bf16(c1, d0, acc10);
      acc11 = wmma_bf16(c1, d1, acc11);
    }
  }

  // Epilogue. C/D layout: VGPR v -> row v + (lane>=16 ? 8 : 0), col = lane&15.
  const int rhalf = (lane >> 4) << 3;
  const int cl    = lane & 15;
  #pragma unroll
  for (int mi = 0; mi < 2; mi++) {
    v8f av0 = mi ? acc10 : acc00;
    v8f av1 = mi ? acc11 : acc01;
    #pragma unroll
    for (int ni = 0; ni < 2; ni++) {
      v8f a = ni ? av1 : av0;
      const int colg = n0 + nsub + ni * 16 + cl;
      if (colg >= Nout) continue;
      const float bv = bias ? bias[colg] : 0.f;
      #pragma unroll
      for (int v = 0; v < 8; v++) {
        const int rowg = m0 + msub + mi * 16 + rhalf + v;
        if (rowg < M) {
          float val = a[v] + bv;
          if (RELU) val = fmaxf(val, 0.f);
          if constexpr (OBF) ((__bf16*)C)[(size_t)rowg * ldc + colg] = (__bf16)val;
          else               ((float*)C)[(size_t)rowg * ldc + colg] = val;
        }
      }
    }
  }
}

// ---------------------------------------------------------------------------
// GEMM (bf16 A): A tiles staged into LDS with GLOBAL_LOAD_ASYNC_TO_LDS_B128
// (no VGPR round-trip, ASYNCcnt tracked). Used for activations stored bf16.
// ---------------------------------------------------------------------------
template<bool RELU, bool OBF>
__global__ void __launch_bounds__(256, 2) gemm_wmma_b(
    const __bf16* __restrict__ A, int lda,
    const __bf16* __restrict__ Wt, const float* __restrict__ bias,
    void* __restrict__ C, int ldc, int M, int K, int Nout)
{
  __shared__ __bf16 As[128 * 32];

  const int tid  = threadIdx.x;
  const int lane = tid & 31;
  const int wid  = tid >> 5;
  const int msub = (wid & 3) << 5;
  const int nsub = (wid >> 2) << 5;
  const int m0   = blockIdx.x * 128;
  const int n0   = blockIdx.y * 64;

  __builtin_prefetch(Wt + (size_t)(n0 + nsub + (lane & 15)) * K, 0, 1);

  v8f acc00 = {}; v8f acc01 = {}; v8f acc10 = {}; v8f acc11 = {};

  for (int k0 = 0; k0 < K; k0 += 32) {
    __syncthreads();
    {
      const int r  = tid >> 1;
      const int hh = (tid & 1) << 4;   // elements (32B half-row)
      int rg = m0 + r; rg = rg < M ? rg : M - 1;
      async_copy32(A + (size_t)rg * lda + k0 + hh, As + r * 32 + hh);
    }
    async_wait();
    __syncthreads();

    v16bf a0 = load_a_frag(As, msub,      32, 0, lane);
    v16bf a1 = load_a_frag(As, msub + 16, 32, 0, lane);
    v16bf b0 = load_b_frag(Wt, K, n0 + nsub,      k0, lane, Nout);
    v16bf b1 = load_b_frag(Wt, K, n0 + nsub + 16, k0, lane, Nout);
    acc00 = wmma_bf16(a0, b0, acc00);
    acc01 = wmma_bf16(a0, b1, acc01);
    acc10 = wmma_bf16(a1, b0, acc10);
    acc11 = wmma_bf16(a1, b1, acc11);
  }

  const int rhalf = (lane >> 4) << 3;
  const int cl    = lane & 15;
  #pragma unroll
  for (int mi = 0; mi < 2; mi++) {
    v8f av0 = mi ? acc10 : acc00;
    v8f av1 = mi ? acc11 : acc01;
    #pragma unroll
    for (int ni = 0; ni < 2; ni++) {
      v8f a = ni ? av1 : av0;
      const int colg = n0 + nsub + ni * 16 + cl;
      if (colg >= Nout) continue;
      const float bv = bias ? bias[colg] : 0.f;
      #pragma unroll
      for (int v = 0; v < 8; v++) {
        const int rowg = m0 + msub + mi * 16 + rhalf + v;
        if (rowg < M) {
          float val = a[v] + bv;
          if (RELU) val = fmaxf(val, 0.f);
          if constexpr (OBF) ((__bf16*)C)[(size_t)rowg * ldc + colg] = (__bf16)val;
          else               ((float*)C)[(size_t)rowg * ldc + colg] = val;
        }
      }
    }
  }
}

// ---------------------------------------------------------------------------
// Fused edge-encoder + scatter:  e = relu(attr@W1+b1)@W2+b2 ; agg_e[dst] += e
// 64 edges/block, t kept in LDS as bf16 (32KB). w1t is [256][32] zero-padded.
// ---------------------------------------------------------------------------
__global__ void __launch_bounds__(256, 2) edge_mlp_scatter(
    const float* __restrict__ eattr, const int* __restrict__ eidx, int E,
    const __bf16* __restrict__ w1t, const float* __restrict__ b1,
    const __bf16* __restrict__ w2t, const float* __restrict__ b2,
    float* __restrict__ agg_e)
{
  __shared__ __bf16 att[64 * 32];
  __shared__ __bf16 tb [64 * 256];
  __shared__ int    dsts[64];

  const int tid  = threadIdx.x;
  const int lane = tid & 31;
  const int wid  = tid >> 5;
  const int e0   = blockIdx.x * 64;

  { // stage edge_attr (pad K 16->32 with zeros) + dst indices
    const int r   = tid >> 2;
    const int seg = (tid & 3) << 2;
    const int eg  = e0 + r;
    float4 f = make_float4(0.f, 0.f, 0.f, 0.f);
    if (eg < E) f = *(const float4*)(eattr + (size_t)eg * 16 + seg);
    union { __bf16 b[4]; uint2 q; } pk;
    pk.b[0]=(__bf16)f.x; pk.b[1]=(__bf16)f.y; pk.b[2]=(__bf16)f.z; pk.b[3]=(__bf16)f.w;
    *(uint2*)(att + r * 32 + seg) = pk.q;
    uint2 zz; zz.x = 0u; zz.y = 0u;
    *(uint2*)(att + r * 32 + 16 + seg) = zz;
    if ((tid & 3) == 0) dsts[r] = (eg < E) ? eidx[(size_t)E + eg] : -1;
  }
  __syncthreads();

  const int mrow  = (wid & 3) << 4;   // 16-row strip per wave
  const int nb    = (wid >> 2) << 7;  // 0 or 128 column half
  const int cl    = lane & 15;
  const int rhalf = (lane >> 4) << 3;

  { // GEMM1: [64,32pad] @ w1t -> relu -> bf16 tile in LDS
    v16bf a = load_a_frag(att, mrow, 32, 0, lane);
    #pragma unroll
    for (int nf = 0; nf < 8; nf++) {
      const int nc = nb + (nf << 4);
      v16bf b = load_b_frag(w1t, 32, nc, 0, lane, 256);
      v8f c = {};
      c = wmma_bf16(a, b, c);
      const int col = nc + cl;
      const float bv = b1[col];
      #pragma unroll
      for (int v = 0; v < 8; v++) {
        const int row = mrow + rhalf + v;
        tb[row * 256 + col] = (__bf16)fmaxf(c[v] + bv, 0.f);
      }
    }
  }
  __syncthreads();

  // GEMM2: [64,256] @ w2t -> +b2 -> atomic scatter into agg_e[dst]
  v8f acc[8] = {};
  for (int k0 = 0; k0 < 256; k0 += 32) {
    v16bf a = load_a_frag(tb, mrow, 256, k0, lane);
    #pragma unroll
    for (int nf = 0; nf < 8; nf++) {
      v16bf b = load_b_frag(w2t, 256, nb + (nf << 4), k0, lane, 256);
      acc[nf] = wmma_bf16(a, b, acc[nf]);
    }
  }
  #pragma unroll
  for (int nf = 0; nf < 8; nf++) {
    const int col = nb + (nf << 4) + cl;
    const float bv = b2[col];
    #pragma unroll
    for (int v = 0; v < 8; v++) {
      const int row = mrow + rhalf + v;
      const int d = dsts[row];
      if (d >= 0) atomicAdd(agg_e + (size_t)d * 256 + col, acc[nf][v] + bv);
    }
  }
}

// ---------------------------------------------------------------------------
// Small helper kernels
// ---------------------------------------------------------------------------
__global__ void wt_transpose(const float* __restrict__ W, __bf16* __restrict__ Wt,
                             int K, int Nn, int Kp) {
  const int idx = blockIdx.x * 256 + threadIdx.x;
  if (idx >= Nn * Kp) return;
  const int n = idx / Kp, k = idx % Kp;
  Wt[idx] = (k < K) ? (__bf16)W[(size_t)k * Nn + n] : (__bf16)0.f;
}

// Wt[n*256+k] = bf16( sum_j Wm[k][j] * Wu[j][n] )  (folded chain, transposed)
__global__ void wt_combine(const float* __restrict__ Wm, const float* __restrict__ Wu,
                           __bf16* __restrict__ Wt) {
  const int n = blockIdx.x;
  const int k = threadIdx.x;
  float acc = 0.f;
  const float* wm = Wm + (size_t)k * 256;
  for (int j = 0; j < 256; j++) acc = fmaf(wm[j], Wu[(size_t)j * 256 + n], acc);
  Wt[(size_t)n * 256 + k] = (__bf16)acc;
}

__global__ void zero_f4(float* __restrict__ p, size_t n4) {
  const size_t i = (size_t)blockIdx.x * 256 + threadIdx.x;
  if (i < n4) ((float4*)p)[i] = make_float4(0.f, 0.f, 0.f, 0.f);
}

__global__ void copy_tail(const float* __restrict__ x, __bf16* __restrict__ xin, int Nn) {
  const int idx = blockIdx.x * 256 + threadIdx.x;
  if (idx >= Nn * 32) return;
  const int n = idx >> 5, j = idx & 31;
  xin[(size_t)n * 64 + 32 + j] = (__bf16)x[(size_t)n * 160 + 128 + j];
}

__global__ void scatter_h(const float* __restrict__ h, const int* __restrict__ eidx,
                          float* __restrict__ agg, int E) {
  const size_t idx = (size_t)blockIdx.x * 256 + threadIdx.x;
  if (idx >= (size_t)E * 64) return;
  const int e = (int)(idx >> 6);
  const int c = (int)(idx & 63) << 2;
  const int s = eidx[e];
  const int d = eidx[(size_t)E + e];
  float4 v = *(const float4*)(h + (size_t)s * 256 + c);
  float* p = agg + (size_t)d * 256 + c;
  atomicAdd(p + 0, v.x); atomicAdd(p + 1, v.y);
  atomicAdd(p + 2, v.z); atomicAdd(p + 3, v.w);
}

// h = layernorm(h + u) * g + b ; one wave per node (8 nodes / 256-thr block)
__global__ void ln_residual(float* __restrict__ h, const float* __restrict__ u,
                            const float* __restrict__ g, const float* __restrict__ b,
                            int Nn) {
  const int lane = threadIdx.x & 31;
  const int node = blockIdx.x * 8 + (threadIdx.x >> 5);
  if (node >= Nn) return;
  float* hp = h + (size_t)node * 256;
  const float* up = u + (size_t)node * 256;
  float v[8]; float s = 0.f;
  #pragma unroll
  for (int i = 0; i < 8; i++) { const int c = lane * 8 + i; v[i] = hp[c] + up[c]; s += v[i]; }
  #pragma unroll
  for (int o = 16; o > 0; o >>= 1) s += __shfl_xor(s, o, 32);
  const float mu = s * (1.f / 256.f);
  float q = 0.f;
  #pragma unroll
  for (int i = 0; i < 8; i++) { const float d = v[i] - mu; q += d * d; }
  #pragma unroll
  for (int o = 16; o > 0; o >>= 1) q += __shfl_xor(q, o, 32);
  const float rs = rsqrtf(q * (1.f / 256.f) + 1e-5f);
  #pragma unroll
  for (int i = 0; i < 8; i++) {
    const int c = lane * 8 + i;
    hp[c] = (v[i] - mu) * rs * g[c] + b[c];
  }
}

__global__ void pool_acc(const float* __restrict__ h, const int* __restrict__ batch,
                         const int* __restrict__ ntype, float* __restrict__ psum,
                         float* __restrict__ cnt, int Nn) {
  const size_t idx = (size_t)blockIdx.x * 256 + threadIdx.x;
  if (idx >= (size_t)Nn * 64) return;
  const int n = (int)(idx >> 6);
  const int c = (int)(idx & 63) << 2;
  const int k = ntype[n], gg = batch[n];
  float4 hv = *(const float4*)(h + (size_t)n * 256 + c);
  float* p = psum + (size_t)gg * 768 + k * 256 + c;
  atomicAdd(p + 0, hv.x); atomicAdd(p + 1, hv.y);
  atomicAdd(p + 2, hv.z); atomicAdd(p + 3, hv.w);
  if ((idx & 63) == 0) atomicAdd(cnt + gg * 3 + k, 1.f);
}

__global__ void pool_final(const float* __restrict__ psum, const float* __restrict__ cnt,
                           float* __restrict__ z, int Bn) {
  const int i = blockIdx.x * 256 + threadIdx.x;
  if (i >= Bn * 768) return;
  const int gg = i / 768;
  const int k  = (i % 768) >> 8;
  z[i] = psum[i] / fmaxf(cnt[gg * 3 + k], 1.f);
}

__global__ void batch_out(const int* __restrict__ batch, float* __restrict__ ob, int Nn) {
  const int i = blockIdx.x * 256 + threadIdx.x;
  if (i < Nn) ob[i] = (float)batch[i];
}

// ---------------------------------------------------------------------------
// Orchestration
// ---------------------------------------------------------------------------
extern "C" void kernel_launch(void* const* d_in, const int* in_sizes, int n_in,
                              void* d_out, int out_size, void* d_ws, size_t ws_size,
                              hipStream_t stream) {
  (void)n_in; (void)out_size; (void)ws_size;
  const float* x      = (const float*)d_in[0];
  const float* eattr  = (const float*)d_in[1];
  const float* geo_w  = (const float*)d_in[2];
  const float* geo_b  = (const float*)d_in[3];
  const float* nin_w1 = (const float*)d_in[4];
  const float* nin_b1 = (const float*)d_in[5];
  const float* nin_w2 = (const float*)d_in[6];
  const float* nin_b2 = (const float*)d_in[7];
  const float* ee_w1  = (const float*)d_in[8];
  const float* ee_b1  = (const float*)d_in[9];
  const float* ee_w2  = (const float*)d_in[10];
  const float* ee_b2  = (const float*)d_in[11];
  const float* msgx_w = (const float*)d_in[12];
  const float* msge_w = (const float*)d_in[13];
  const float* upd_w1 = (const float*)d_in[14];
  const float* upd_b1 = (const float*)d_in[15];
  const float* upd_w2 = (const float*)d_in[16];
  const float* upd_b2 = (const float*)d_in[17];
  const float* ln_g   = (const float*)d_in[18];
  const float* ln_b   = (const float*)d_in[19];
  const int*   eidx   = (const int*)d_in[20];
  const int*   batch  = (const int*)d_in[21];
  const int*   ntype  = (const int*)d_in[22];

  const int N  = in_sizes[0] / 160;
  const int E  = in_sizes[1] / 16;
  const int Bn = 8;
  const int H  = 256;
  const int L  = 4;

  // Output layout: z[B,768] | h[N,256] | batch[N] (as float)
  float* out = (float*)d_out;
  float* z   = out;
  float* h   = out + (size_t)Bn * 768;
  float* ob  = h + (size_t)N * H;

  // Workspace carve-up
  char* w = (char*)d_ws;
  auto alloc = [&](size_t bytes) -> char* {
    char* p = w; w += (bytes + 255) & ~(size_t)255; return p;
  };
  __bf16* xinb    = (__bf16*)alloc((size_t)N * 64 * 2);   // bf16 activations
  __bf16* tmpAb   = (__bf16*)alloc((size_t)N * H * 2);    // bf16 relu outputs
  float*  tmpB    = (float*)alloc((size_t)N * H * 4);
  float*  agge    = (float*)alloc((size_t)N * H * 4);
  float*  aggh    = (float*)alloc((size_t)N * H * 4);
  __bf16* geo_wt  = (__bf16*)alloc((size_t)32 * 128 * 2);
  __bf16* nin_w1t = (__bf16*)alloc((size_t)256 * 64 * 2);
  __bf16* nin_w2t = (__bf16*)alloc((size_t)256 * 256 * 2);
  __bf16* ee_w1t  = (__bf16*)alloc((size_t)256 * 32 * 2);
  __bf16* ee_w2t  = (__bf16*)alloc((size_t)256 * 256 * 2);
  __bf16* w1xt    = (__bf16*)alloc((size_t)L * 256 * 256 * 2);
  __bf16* w1et    = (__bf16*)alloc((size_t)L * 256 * 256 * 2);
  __bf16* w2t     = (__bf16*)alloc((size_t)L * 256 * 256 * 2);
  float*  psum    = (float*)alloc(((size_t)Bn * 768 + 32) * 4);
  float*  cnt     = psum + (size_t)Bn * 768;

  // --- Weight prep: transpose->bf16, and fold msg*upd_w1 chains ---
  auto tc = [&](const float* W, __bf16* Wt, int K, int Nn2, int Kp) {
    const int tot = Nn2 * Kp;
    wt_transpose<<<(tot + 255) / 256, 256, 0, stream>>>(W, Wt, K, Nn2, Kp);
  };
  tc(geo_w,  geo_wt,  128, 32, 128);
  tc(nin_w1, nin_w1t, 64, 256, 64);
  tc(nin_w2, nin_w2t, 256, 256, 256);
  tc(ee_w1,  ee_w1t,  16, 256, 32);      // zero-pad K 16 -> 32
  tc(ee_w2,  ee_w2t,  256, 256, 256);
  for (int l = 0; l < L; l++) {
    tc(upd_w2 + (size_t)l * 65536, w2t + (size_t)l * 65536, 256, 256, 256);
    wt_combine<<<256, 256, 0, stream>>>(msgx_w + (size_t)l * 65536,
                                        upd_w1 + (size_t)l * 65536,
                                        w1xt + (size_t)l * 65536);
    wt_combine<<<256, 256, 0, stream>>>(msge_w + (size_t)l * 65536,
                                        upd_w1 + (size_t)l * 65536,
                                        w1et + (size_t)l * 65536);
  }

  const int gmx = (N + 127) / 128;

  // --- Node front: geo proj (bf16 out) + concat + node_in MLP -> h ---
  gemm_wmma<true, false, true><<<dim3(gmx, 1), 256, 0, stream>>>(
      x, nullptr, 160, geo_wt, nullptr, geo_b, xinb, 64, N, 128, 32);
  copy_tail<<<(N * 32 + 255) / 256, 256, 0, stream>>>(x, xinb, N);
  gemm_wmma_b<true, true><<<dim3(gmx, 4), 256, 0, stream>>>(
      xinb, 64, nin_w1t, nin_b1, tmpAb, 256, N, 64, 256);
  gemm_wmma_b<false, false><<<dim3(gmx, 4), 256, 0, stream>>>(
      tmpAb, 256, nin_w2t, nin_b2, h, 256, N, 256, 256);

  // --- Edge encoder fused with scatter into agg_e (once, layer-invariant) ---
  const size_t nh4 = (size_t)N * H / 4;
  zero_f4<<<(unsigned)((nh4 + 255) / 256), 256, 0, stream>>>(agge, nh4);
  edge_mlp_scatter<<<(E + 63) / 64, 256, 0, stream>>>(
      eattr, eidx, E, ee_w1t, ee_b1, ee_w2t, ee_b2, agge);

  // --- Message-passing layers (GEMMs moved from E-space to N-space) ---
  for (int l = 0; l < L; l++) {
    zero_f4<<<(unsigned)((nh4 + 255) / 256), 256, 0, stream>>>(aggh, nh4);
    const size_t sc = (size_t)E * 64;
    scatter_h<<<(unsigned)((sc + 255) / 256), 256, 0, stream>>>(h, eidx, aggh, E);
    gemm_wmma<true, true, true><<<dim3(gmx, 4), 256, 0, stream>>>(
        aggh, agge, 256, w1xt + (size_t)l * 65536, w1et + (size_t)l * 65536,
        upd_b1 + (size_t)l * 256, tmpAb, 256, N, 256, 256);
    gemm_wmma_b<false, false><<<dim3(gmx, 4), 256, 0, stream>>>(
        tmpAb, 256, w2t + (size_t)l * 65536, upd_b2 + (size_t)l * 256,
        tmpB, 256, N, 256, 256);
    ln_residual<<<(N + 7) / 8, 256, 0, stream>>>(
        h, tmpB, ln_g + (size_t)l * 256, ln_b + (size_t)l * 256, N);
  }

  // --- Typed mean-pool per graph + outputs ---
  const size_t pz4 = ((size_t)Bn * 768 + 32) / 4;
  zero_f4<<<(unsigned)((pz4 + 255) / 256), 256, 0, stream>>>(psum, pz4);
  const size_t pa = (size_t)N * 64;
  pool_acc<<<(unsigned)((pa + 255) / 256), 256, 0, stream>>>(h, batch, ntype, psum, cnt, N);
  pool_final<<<(Bn * 768 + 255) / 256, 256, 0, stream>>>(psum, cnt, z, Bn);
  batch_out<<<(N + 255) / 256, 256, 0, stream>>>(batch, ob, N);
}
